// BilstmCrfModel_16578573762648
// MI455X (gfx1250) — matrix operand order repaired and checked by reading, not compiled, not run on hardware
//
#include <hip/hip_runtime.h>

// ---------------- problem constants ----------------
#define BB    64
#define SS    512
#define EMB   256
#define HH    256      // H = HID/2
#define GG    1024     // 4*H gates
#define HID   512
#define NL    17
#define NLP   32       // padded label count for WMMA N
#define VOCAB 20000

// ---------------- WMMA types (CDNA5, wave32) ----------------
typedef __attribute__((ext_vector_type(16))) __bf16 v16bf;
typedef __attribute__((ext_vector_type(8)))  float  v8f;

union ABFrag { v16bf v; uint4 q[2]; };   // 32 bytes = 16 bf16

static __device__ __forceinline__ float sigm_(float x) { return 1.0f / (1.0f + __expf(-x)); }

// =====================================================================
// Prep: bf16 conversions, weight transposes, fused biases, padding.
//   wihT/whhT : [K=256][N=1024]  (B-matrix: lane = N col, contiguous N)
//   woutT     : [K=512][N=32]    rows >=17 zero
// =====================================================================
__global__ void prep_kernel(
    const float* __restrict__ emb,
    const float* __restrict__ wihf, const float* __restrict__ whhf,
    const float* __restrict__ bihf, const float* __restrict__ bhhf,
    const float* __restrict__ wihb, const float* __restrict__ whhb,
    const float* __restrict__ bihb, const float* __restrict__ bhhb,
    const float* __restrict__ wout, const float* __restrict__ bout,
    __bf16* embB, __bf16* wihTf, __bf16* whhTf, __bf16* wihTb, __bf16* whhTb,
    float* biasf, float* biasb, __bf16* woutT, float* boutP)
{
    const long stride = (long)gridDim.x * blockDim.x;
    const long i0     = (long)blockIdx.x * blockDim.x + threadIdx.x;

    for (long t = i0; t < (long)VOCAB * EMB; t += stride)
        embB[t] = (__bf16)emb[t];

    for (long t = i0; t < (long)EMB * GG; t += stride) {   // EMB == HH == 256
        long k = t / GG, n = t % GG;
        wihTf[t] = (__bf16)wihf[n * EMB + k];
        wihTb[t] = (__bf16)wihb[n * EMB + k];
        whhTf[t] = (__bf16)whhf[n * HH  + k];
        whhTb[t] = (__bf16)whhb[n * HH  + k];
    }
    for (long t = i0; t < GG; t += stride) {
        biasf[t] = bihf[t] + bhhf[t];
        biasb[t] = bihb[t] + bhhb[t];
    }
    for (long t = i0; t < (long)HID * NLP; t += stride) {
        long k = t / NLP, n = t % NLP;
        woutT[t] = (n < NL) ? (__bf16)wout[n * HID + k] : (__bf16)0.0f;
    }
    for (long t = i0; t < NLP; t += stride)
        boutP[t] = (t < NL) ? bout[t] : 0.0f;
}

// =====================================================================
// Kernel A: xp = emb[chars] @ w_ih^T + (b_ih + b_hh)
//   M = B*S = 32768 (row tiles of 16), N = 1024, K = 256.
//   grid (2048, 8), block 256 (8 waves, 16 N-cols each). Launched per dir.
// =====================================================================
__global__ __launch_bounds__(256) void xproj_kernel(
    const int* __restrict__ chars, const __bf16* __restrict__ embB,
    const __bf16* __restrict__ wihT, const float* __restrict__ bias,
    float* __restrict__ xp)
{
    const int lane = threadIdx.x & 31;
    const int wave = threadIdx.x >> 5;
    const int rt   = blockIdx.x;
    const int n0   = (blockIdx.y * 8 + wave) * 16;
    const int lh   = lane & 15;
    const int hi   = (lane >> 4) & 1;

    const int   arow  = chars[rt * 16 + lh];           // gather row
    const __bf16* abase = embB + (long)arow * EMB;

    const float bsum = bias[n0 + lh];
    v8f acc;
#pragma unroll
    for (int r = 0; r < 8; ++r) acc[r] = bsum;

#pragma unroll
    for (int kc = 0; kc < EMB / 32; ++kc) {
        const int k0 = kc * 32 + (hi ? 8 : 0);
        ABFrag a;
        a.q[0] = *(const uint4*)(abase + k0);
        a.q[1] = *(const uint4*)(abase + k0 + 16);
        const __bf16* bb = wihT + (long)(kc * 32 + (hi ? 8 : 0)) * GG + n0 + lh;
        ABFrag b;
#pragma unroll
        for (int s = 0; s < 8; ++s) { b.v[s] = bb[(long)s * GG]; b.v[8 + s] = bb[(long)(16 + s) * GG]; }
        acc = __builtin_amdgcn_wmma_f32_16x16x32_bf16(false, a.v, false, b.v,
                                                      (short)0, acc, false, false);
    }
    const int nn = n0 + lh;
#pragma unroll
    for (int r = 0; r < 8; ++r)
        xp[(long)(rt * 16 + r + hi * 8) * GG + nn] = acc[r];
}

// =====================================================================
// Kernel B: the LSTM recurrence. grid = 2 (dir), block = 1024 (32 waves).
//   Per step: gates(64x1024) = xp[:,t,:] + h(64x256) @ whhT(256x1024)
//   Wave w: mt = w&3 (batch rows), owns hidden-col tiles {2*(w>>2), +1},
//   i.e. gate tiles at N = gate*256 + jt*16 -> i/f/g/o register-local.
//   h kept in LDS as bf16 (WMMA A source); c in VGPRs.
// =====================================================================
__global__ __launch_bounds__(1024) void lstm_kernel(
    const float* __restrict__ xpf, const float* __restrict__ xpb,
    const __bf16* __restrict__ whhTf, const __bf16* __restrict__ whhTb,
    __bf16* __restrict__ hsf, __bf16* __restrict__ hsb)
{
    __shared__ __bf16 hbuf[BB * HH];                   // 32 KB

    const int dir = blockIdx.x;
    const float*  xp   = dir ? xpb   : xpf;
    const __bf16* whhT = dir ? whhTb : whhTf;
    __bf16*       hs   = dir ? hsb   : hsf;
    const int     rev  = dir;

    const int tid  = threadIdx.x;
    const int lane = tid & 31;
    const int wave = tid >> 5;
    const int mt   = wave & 3;      // batch row tile
    const int jg   = wave >> 2;     // hidden col group (0..7) -> jt in {2jg,2jg+1}
    const int lh   = lane & 15;
    const int hi   = (lane >> 4) & 1;

    for (int i = tid; i < BB * HH / 2; i += 1024) ((unsigned int*)hbuf)[i] = 0u;
    __syncthreads();

    v8f c[2];
#pragma unroll
    for (int g = 0; g < 2; ++g)
#pragma unroll
        for (int r = 0; r < 8; ++r) c[g][r] = 0.0f;

    for (int step = 0; step < SS; ++step) {
        const int t = rev ? (SS - 1 - step) : step;

        // ---- init accumulators from precomputed xp (includes biases) ----
        v8f acc[8];
#pragma unroll
        for (int g = 0; g < 2; ++g) {
            const int jt = jg * 2 + g;
#pragma unroll
            for (int gate = 0; gate < 4; ++gate) {
                const int nn = gate * HH + jt * 16 + lh;
                v8f a;
#pragma unroll
                for (int r = 0; r < 8; ++r)
                    a[r] = xp[((long)(mt * 16 + r + hi * 8) * SS + t) * GG + nn];
                acc[g * 4 + gate] = a;
            }
        }
        if (step + 1 < SS) {   // warm L2/L0 for next step's xp slice
            const int tn = rev ? (SS - 2 - step) : (step + 1);
            __builtin_prefetch(xp + ((long)(mt * 16 + lh) * SS + tn) * GG + jg * 128, 0, 0);
        }

        // ---- h_prev @ whhT via WMMA (K = 256, 8 chunks of 32) ----
#pragma unroll
        for (int kc = 0; kc < HH / 32; ++kc) {
            const int k0 = kc * 32 + (hi ? 8 : 0);
            const __bf16* ab = hbuf + (mt * 16 + lh) * HH;
            ABFrag a;
            a.q[0] = *(const uint4*)(ab + k0);
            a.q[1] = *(const uint4*)(ab + k0 + 16);
            const __bf16* brow = whhT + (long)(kc * 32 + (hi ? 8 : 0)) * GG;
#pragma unroll
            for (int g = 0; g < 2; ++g) {
                const int jt = jg * 2 + g;
#pragma unroll
                for (int gate = 0; gate < 4; ++gate) {
                    const __bf16* bb = brow + gate * HH + jt * 16 + lh;
                    ABFrag b;
#pragma unroll
                    for (int s = 0; s < 8; ++s) { b.v[s] = bb[(long)s * GG]; b.v[8 + s] = bb[(long)(16 + s) * GG]; }
                    acc[g * 4 + gate] = __builtin_amdgcn_wmma_f32_16x16x32_bf16(
                        false, a.v, false, b.v, (short)0, acc[g * 4 + gate], false, false);
                }
            }
        }
        __syncthreads();   // all reads of hbuf done before overwrite

        // ---- gate nonlinearities, c/h update, publish h ----
#pragma unroll
        for (int g = 0; g < 2; ++g) {
            const int jt = jg * 2 + g;
            const v8f iv = acc[g * 4 + 0], fv = acc[g * 4 + 1];
            const v8f gv = acc[g * 4 + 2], ov = acc[g * 4 + 3];
#pragma unroll
            for (int r = 0; r < 8; ++r) {
                float cc = sigm_(fv[r]) * c[g][r] + sigm_(iv[r]) * tanhf(gv[r]);
                float hh = sigm_(ov[r]) * tanhf(cc);
                c[g][r] = cc;
                const int brow = mt * 16 + r + hi * 8;
                const int col  = jt * 16 + lh;
                const __bf16 hb = (__bf16)hh;
                hbuf[brow * HH + col] = hb;
                hs[((long)brow * SS + t) * HH + col] = hb;
            }
        }
        __syncthreads();   // new h visible before next step's reads
    }
}

// =====================================================================
// Kernel C: emissions = [hf|hb] @ w_out^T + b_out   (N padded to 32)
//   grid 2048, block 64 (2 waves, 16 cols each), K = 512.
// =====================================================================
__global__ __launch_bounds__(64) void emis_kernel(
    const __bf16* __restrict__ hsf, const __bf16* __restrict__ hsb,
    const __bf16* __restrict__ woutT, const float* __restrict__ boutP,
    float* __restrict__ emis)
{
    const int lane = threadIdx.x & 31;
    const int wave = threadIdx.x >> 5;
    const int rt   = blockIdx.x;
    const int n0   = wave * 16;
    const int lh   = lane & 15;
    const int hi   = (lane >> 4) & 1;
    const long row = (long)rt * 16 + lh;               // flattened b*S+t

    v8f acc;
    const float bsum = boutP[n0 + lh];
#pragma unroll
    for (int r = 0; r < 8; ++r) acc[r] = bsum;

#pragma unroll
    for (int kc = 0; kc < HID / 32; ++kc) {
        const int k0 = kc * 32 + (hi ? 8 : 0);
        const __bf16* src = (k0 < HH) ? (hsf + row * HH + k0)
                                      : (hsb + row * HH + (k0 - HH));
        ABFrag a;
        a.q[0] = *(const uint4*)(src);
        a.q[1] = *(const uint4*)(src + 16);
        const __bf16* bb = woutT + (long)(kc * 32 + (hi ? 8 : 0)) * NLP + n0 + lh;
        ABFrag b;
#pragma unroll
        for (int s = 0; s < 8; ++s) { b.v[s] = bb[(long)s * NLP]; b.v[8 + s] = bb[(long)(16 + s) * NLP]; }
        acc = __builtin_amdgcn_wmma_f32_16x16x32_bf16(false, a.v, false, b.v,
                                                      (short)0, acc, false, false);
    }
#pragma unroll
    for (int r = 0; r < 8; ++r)
        emis[(long)(rt * 16 + r + hi * 8) * NLP + (n0 + lh)] = acc[r];
}

// =====================================================================
// Kernel D: CRF NLL per batch row. One wave per row; lane j = tag j.
//   (chars are in [1,VOCAB) so mask is all-true, seq_end = S-1.)
// =====================================================================
__global__ __launch_bounds__(32) void crf_kernel(
    const float* __restrict__ emis, const int* __restrict__ labels,
    const float* __restrict__ startT, const float* __restrict__ endT,
    const float* __restrict__ trans, float* __restrict__ partial)
{
    const int b = blockIdx.x;
    const int j = threadIdx.x;
    const float* em  = emis + (long)b * SS * NLP;
    const int*   lab = labels + (long)b * SS;

    // gold path score (parallel over t, wave-reduced)
    float sc = 0.0f;
    for (int t = 1 + j; t < SS; t += 32) {
        const int lp = lab[t - 1], lt = lab[t];
        sc += trans[lp * NL + lt] + em[t * NLP + lt];
    }
#pragma unroll
    for (int o = 16; o > 0; o >>= 1) sc += __shfl_down(sc, o);
    float score = __shfl(sc, 0);
    const int l0 = lab[0], lend = lab[SS - 1];
    score += startT[l0] + em[l0] + endT[lend];

    // forward algorithm (logsumexp recurrence)
    const bool act = (j < NL);
    float tcol[NL];
#pragma unroll
    for (int i = 0; i < NL; ++i) tcol[i] = act ? trans[i * NL + j] : 0.0f;
    float alpha = act ? (startT[j] + em[j]) : -3.0e38f;

    for (int t = 1; t < SS; ++t) {
        float v[NL];
        float m = -3.0e38f;
#pragma unroll
        for (int i = 0; i < NL; ++i) {
            const float ai = __shfl(alpha, i);
            v[i] = ai + tcol[i];
            m = fmaxf(m, v[i]);
        }
        float s = 0.0f;
#pragma unroll
        for (int i = 0; i < NL; ++i) s += __expf(v[i] - m);
        const float na = m + __logf(s) + (act ? em[t * NLP + j] : 0.0f);
        alpha = act ? na : -3.0e38f;
    }

    float fv = act ? (alpha + endT[j]) : -3.0e38f;
    float mm = fv;
#pragma unroll
    for (int o = 16; o > 0; o >>= 1) mm = fmaxf(mm, __shfl_xor(mm, o));
    float es = __expf(fv - mm);
#pragma unroll
    for (int o = 16; o > 0; o >>= 1) es += __shfl_xor(es, o);
    const float denom = mm + __logf(es);
    if (j == 0) partial[b] = denom - score;
}

__global__ __launch_bounds__(32) void reduce_kernel(const float* __restrict__ partial,
                                                    float* __restrict__ out)
{
    const int j = threadIdx.x;
    float s = partial[j] + partial[j + 32];
#pragma unroll
    for (int o = 16; o > 0; o >>= 1) s += __shfl_down(s, o);
    if (j == 0) out[0] = s;
}

// =====================================================================
extern "C" void kernel_launch(void* const* d_in, const int* in_sizes, int n_in,
                              void* d_out, int out_size, void* d_ws, size_t ws_size,
                              hipStream_t stream)
{
    const int*   chars  = (const int*)d_in[0];
    const int*   labels = (const int*)d_in[1];
    const float* emb    = (const float*)d_in[2];
    const float* wihf   = (const float*)d_in[3];
    const float* whhf   = (const float*)d_in[4];
    const float* bihf   = (const float*)d_in[5];
    const float* bhhf   = (const float*)d_in[6];
    const float* wihb   = (const float*)d_in[7];
    const float* whhb   = (const float*)d_in[8];
    const float* bihb   = (const float*)d_in[9];
    const float* bhhb   = (const float*)d_in[10];
    const float* wout   = (const float*)d_in[11];
    const float* bout   = (const float*)d_in[12];
    const float* startT = (const float*)d_in[13];
    const float* endT   = (const float*)d_in[14];
    const float* trans  = (const float*)d_in[15];
    float* out = (float*)d_out;

    char*  ws  = (char*)d_ws;
    size_t off = 0;
    auto alloc = [&](size_t bytes) -> char* {
        char* p = ws + off;
        off = (off + bytes + 255) & ~(size_t)255;
        return p;
    };
    __bf16* embB   = (__bf16*)alloc((size_t)VOCAB * EMB * sizeof(__bf16));
    __bf16* wihTf  = (__bf16*)alloc((size_t)EMB * GG * sizeof(__bf16));
    __bf16* wihTb  = (__bf16*)alloc((size_t)EMB * GG * sizeof(__bf16));
    __bf16* whhTf  = (__bf16*)alloc((size_t)HH  * GG * sizeof(__bf16));
    __bf16* whhTb  = (__bf16*)alloc((size_t)HH  * GG * sizeof(__bf16));
    float*  biasf  = (float*) alloc((size_t)GG * sizeof(float));
    float*  biasb  = (float*) alloc((size_t)GG * sizeof(float));
    __bf16* woutT  = (__bf16*)alloc((size_t)HID * NLP * sizeof(__bf16));
    float*  boutP  = (float*) alloc((size_t)NLP * sizeof(float));
    float*  xpf    = (float*) alloc((size_t)BB * SS * GG * sizeof(float));
    float*  xpb    = (float*) alloc((size_t)BB * SS * GG * sizeof(float));
    __bf16* hsf    = (__bf16*)alloc((size_t)BB * SS * HH * sizeof(__bf16));
    __bf16* hsb    = (__bf16*)alloc((size_t)BB * SS * HH * sizeof(__bf16));
    float*  emis   = (float*) alloc((size_t)BB * SS * NLP * sizeof(float));
    float*  part   = (float*) alloc((size_t)BB * sizeof(float));

    prep_kernel<<<2048, 256, 0, stream>>>(emb, wihf, whhf, bihf, bhhf,
                                          wihb, whhb, bihb, bhhb, wout, bout,
                                          embB, wihTf, whhTf, wihTb, whhTb,
                                          biasf, biasb, woutT, boutP);

    dim3 gA(BB * SS / 16, GG / 128);
    xproj_kernel<<<gA, 256, 0, stream>>>(chars, embB, wihTf, biasf, xpf);
    xproj_kernel<<<gA, 256, 0, stream>>>(chars, embB, wihTb, biasb, xpb);

    lstm_kernel<<<2, 1024, 0, stream>>>(xpf, xpb, whhTf, whhTb, hsf, hsb);

    emis_kernel<<<BB * SS / 16, 64, 0, stream>>>(hsf, hsb, woutT, boutP, emis);

    crf_kernel<<<BB, 32, 0, stream>>>(emis, labels, startT, endT, trans, part);

    reduce_kernel<<<1, 32, 0, stream>>>(part, out);
}
